// LocalCausalSelfAttention_6442450944393
// MI455X (gfx1250) — matrix operand order repaired and checked by reading, not compiled, not run on hardware
//
#include <hip/hip_runtime.h>

// ---------------------------------------------------------------------------
// LocalCausalSelfAttention for MI455X (gfx1250, wave32, WMMA f16 16x16x32)
// ---------------------------------------------------------------------------
typedef __attribute__((ext_vector_type(16))) _Float16 v16h;
typedef __attribute__((ext_vector_type(8)))  float    v8f;

#define WMMA_F16(a, b, c) \
  __builtin_amdgcn_wmma_f32_16x16x32_f16(false, (a), false, (b), (short)0, (c), false, false)

constexpr int Bsz  = 2;
constexpr int S    = 2048;
constexpr int Dm   = 1024;
constexpr int H    = 16;
constexpr int HD   = 64;       // head dim
constexpr int Wwin = 256;      // local window
constexpr int Mrows = Bsz * S; // 4096 token rows
constexpr int N3    = 3 * Dm;  // 3072 qkv cols
constexpr float NEGB = -1e9f;

// ---- WMMA fragment loaders (wave32 layouts, ISA 7.12.2) --------------------
// A 16x32 (f16): lane L holds row M=L&15; halfs map to K:
//   L<16 : halfs 0-7 -> K0..7,  halfs 8-15 -> K16..23
//   L>=16: halfs 0-7 -> K8..15, halfs 8-15 -> K24..31
__device__ __forceinline__ v16h load_A16x32(const _Float16* __restrict__ src,
                                            int ld, int row0, int k0) {
  const int lane = threadIdx.x & 31;
  const int m = lane & 15, hi = lane >> 4;
  const _Float16* p = src + (size_t)(row0 + m) * ld + k0 + hi * 8;
  v16h a;
#pragma unroll
  for (int t = 0; t < 8; ++t) a[t] = p[t];
#pragma unroll
  for (int t = 0; t < 8; ++t) a[8 + t] = p[16 + t];
  return a;
}

// B 32x16 (f16) from a [N][K]-contiguous source (i.e. B^T row-major):
// lane L holds column N=L&15; K mapping matches A: L<16 K0..7,K16..23; L>=16 K8..15,K24..31.
__device__ __forceinline__ v16h load_BT16(const _Float16* __restrict__ src,
                                          int ld, int n0, int k0) {
  const int lane = threadIdx.x & 31;
  const int n = lane & 15, hi = lane >> 4;
  const _Float16* p = src + (size_t)(n0 + n) * ld + k0 + hi * 8;
  v16h b;
#pragma unroll
  for (int t = 0; t < 8; ++t) b[t] = p[t];
#pragma unroll
  for (int t = 0; t < 8; ++t) b[8 + t] = p[16 + t];
  return b;
}
// C/D 16x16 (f32, 8 VGPR): element r -> row = r + 8*(lane>=16), col = lane&15.

// ---- precision-conversion kernels ------------------------------------------
__global__ void k_f32_to_f16(const float* __restrict__ in,
                             _Float16* __restrict__ out, int n) {
  int i = blockIdx.x * 256 + threadIdx.x;
  if (i < n) out[i] = (_Float16)in[i];
}

// in [rows][cols] f32 -> out [cols][rows] f16  (weight transpose for B-frags)
__global__ void k_transpose_f16(const float* __restrict__ in,
                                _Float16* __restrict__ out, int rows, int cols) {
  int i = blockIdx.x * 256 + threadIdx.x;
  if (i < rows * cols) {
    int r = i / cols, c = i - r * cols;
    out[(size_t)c * rows + r] = (_Float16)in[i];
  }
}

// ---- QKV projection: [4096,1024] @ [1024,3072] + bias ----------------------
// One wave computes a 32x64 tile (2 A frags x 4 B frags -> 8 WMMA / k-step);
// epilogue scatters into head layouts:
//   Q -> Qh[b][h][s][hd] (pre-scaled by 1/sqrt(HD))
//   K -> Kh[b][h][s][hd]
//   V -> Vt[b][h][hd][s] (transposed so PV B-frag loads are contiguous)
__global__ __launch_bounds__(32) void k_qkv_gemm(
    const _Float16* __restrict__ A, const _Float16* __restrict__ BT,
    const float* __restrict__ bias, _Float16* __restrict__ Qh,
    _Float16* __restrict__ Kh, _Float16* __restrict__ Vt) {
  const int m0 = blockIdx.x * 32;
  const int n0 = blockIdx.y * 64;
  const int lane = threadIdx.x & 31;
  const int nl = lane & 15, hi = lane >> 4;

  v8f acc[2][4] = {};
  for (int k0 = 0; k0 < Dm; k0 += 32) {
    // stream-ahead hints for the next k block (global_prefetch_b8)
    __builtin_prefetch(A + (size_t)(m0 + nl) * Dm + k0 + 64, 0, 1);
    __builtin_prefetch(BT + (size_t)(n0 + nl) * Dm + k0 + 64, 0, 1);
    v16h a0 = load_A16x32(A, Dm, m0, k0);
    v16h a1 = load_A16x32(A, Dm, m0 + 16, k0);
#pragma unroll
    for (int j = 0; j < 4; ++j) {
      v16h b = load_BT16(BT, Dm, n0 + j * 16, k0);
      acc[0][j] = WMMA_F16(a0, b, acc[0][j]);
      acc[1][j] = WMMA_F16(a1, b, acc[1][j]);
    }
  }
#pragma unroll
  for (int i = 0; i < 2; ++i) {
#pragma unroll
    for (int j = 0; j < 4; ++j) {
#pragma unroll
      for (int r = 0; r < 8; ++r) {
        const int ng = n0 + j * 16 + nl;           // 0..3071
        const int mg = m0 + i * 16 + r + hi * 8;   // 0..4095
        float v = acc[i][j][r] + bias[ng];
        const int sel = ng >> 10;                  // 0=Q 1=K 2=V
        const int d = ng & 1023;
        const int head = d >> 6, hd = d & 63;
        const int b = mg >> 11, s = mg & 2047;
        if (sel == 0)
          Qh[(((size_t)(b * H + head)) * S + s) * HD + hd] = (_Float16)(v * 0.125f);
        else if (sel == 1)
          Kh[(((size_t)(b * H + head)) * S + s) * HD + hd] = (_Float16)v;
        else
          Vt[(((size_t)(b * H + head)) * HD + hd) * S + s] = (_Float16)v;
      }
    }
  }
}

// ---- banded flash attention: one wave per (b, h, 16-query tile) ------------
__global__ __launch_bounds__(32) void k_attn(
    const _Float16* __restrict__ Qh, const _Float16* __restrict__ Kh,
    const _Float16* __restrict__ Vt, _Float16* __restrict__ Mix) {
  __shared__ _Float16 Pl[16 * 32];               // P tile bounce: C-layout -> A-layout

  const int qtiles = S / 16;                     // 128
  const int bh = blockIdx.x / qtiles;
  const int q0 = (blockIdx.x % qtiles) * 16;
  const int b = bh / H, h = bh % H;
  const _Float16* Qp = Qh + (size_t)bh * S * HD;
  const _Float16* Kp = Kh + (size_t)bh * S * HD;
  const _Float16* Vp = Vt + (size_t)bh * HD * S;
  const int lane = threadIdx.x & 31;
  const int nl = lane & 15, hi = lane >> 4;

  const v16h aq0 = load_A16x32(Qp, HD, q0, 0);   // Q already scaled by 1/8
  const v16h aq1 = load_A16x32(Qp, HD, q0, 32);

  float mrow[8], lrow[8];
#pragma unroll
  for (int r = 0; r < 8; ++r) { mrow[r] = -3.0e38f; lrow[r] = 0.0f; }
  v8f o[4] = {};

  int lo = q0 - (Wwin - 1);
  if (lo < 0) lo = 0;
  lo &= ~31;                                     // 32-key aligned start of band
  for (int key0 = lo; key0 < q0 + 16; key0 += 32) {
    // scores: two 16(q)x16(key) C tiles, contraction over hd=64 in 2 steps
    v8f cs[2] = {};
#pragma unroll
    for (int nt = 0; nt < 2; ++nt) {
      cs[nt] = WMMA_F16(aq0, load_BT16(Kp, HD, key0 + nt * 16, 0),  cs[nt]);
      cs[nt] = WMMA_F16(aq1, load_BT16(Kp, HD, key0 + nt * 16, 32), cs[nt]);
    }
    // additive band bias (matches reference 0 / -1e9) + per-row step max
    float sm[8];
#pragma unroll
    for (int r = 0; r < 8; ++r) {
      const int row = q0 + r + hi * 8;
      float best = -3.0e38f;
#pragma unroll
      for (int nt = 0; nt < 2; ++nt) {
        const int col = key0 + nt * 16 + nl;
        float sc = cs[nt][r];
        if (col > row || col < row - (Wwin - 1)) sc += NEGB;
        cs[nt][r] = sc;
        best = fmaxf(best, sc);
      }
      sm[r] = best;
    }
#pragma unroll
    for (int off = 1; off < 16; off <<= 1) {
#pragma unroll
      for (int r = 0; r < 8; ++r)
        sm[r] = fmaxf(sm[r], __shfl_xor(sm[r], off, 32));
    }
    __syncthreads();                              // LDS reuse fence (1 wave)
    // online softmax update + stage P tile (f16) in LDS
    float rs[8];
#pragma unroll
    for (int r = 0; r < 8; ++r) {
      const float mn = fmaxf(mrow[r], sm[r]);
      const float scale = __expf(mrow[r] - mn);
      mrow[r] = mn;
      float part = 0.0f;
#pragma unroll
      for (int nt = 0; nt < 2; ++nt) {
        const float p = __expf(cs[nt][r] - mn);
        part += p;
        Pl[(r + hi * 8) * 32 + nt * 16 + nl] = (_Float16)p;
      }
      rs[r] = part;
      lrow[r] = lrow[r] * scale;
#pragma unroll
      for (int j = 0; j < 4; ++j) o[j][r] = o[j][r] * scale;
    }
#pragma unroll
    for (int off = 1; off < 16; off <<= 1) {
#pragma unroll
      for (int r = 0; r < 8; ++r) rs[r] += __shfl_xor(rs[r], off, 32);
    }
#pragma unroll
    for (int r = 0; r < 8; ++r) lrow[r] += rs[r];
    __syncthreads();
    const v16h pA = load_A16x32(Pl, 32, 0, 0);    // P as 16x32 A-frag (DS loads)
    // O[16 x 64] += P[16 x 32] @ V[32 x 16] per hd tile; Vt gives contiguous B
#pragma unroll
    for (int j = 0; j < 4; ++j)
      o[j] = WMMA_F16(pA, load_BT16(Vp, S, j * 16, key0), o[j]);
  }
  // normalize and write mixed (head-concatenated) as f16 for the output GEMM
#pragma unroll
  for (int j = 0; j < 4; ++j) {
#pragma unroll
    for (int r = 0; r < 8; ++r) {
      const int srow = q0 + r + hi * 8;
      const int col = h * HD + j * 16 + nl;
      Mix[((size_t)(b * S + srow)) * Dm + col] = (_Float16)(o[j][r] / lrow[r]);
    }
  }
}

// ---- output projection: [4096,1024] @ [1024,1024] + bias -> f32 ------------
__global__ __launch_bounds__(32) void k_out_gemm(
    const _Float16* __restrict__ A, const _Float16* __restrict__ BT,
    const float* __restrict__ bias, float* __restrict__ Out) {
  const int m0 = blockIdx.x * 32;
  const int n0 = blockIdx.y * 64;
  const int lane = threadIdx.x & 31;
  const int nl = lane & 15, hi = lane >> 4;

  v8f acc[2][4] = {};
  for (int k0 = 0; k0 < Dm; k0 += 32) {
    __builtin_prefetch(A + (size_t)(m0 + nl) * Dm + k0 + 64, 0, 1);
    __builtin_prefetch(BT + (size_t)(n0 + nl) * Dm + k0 + 64, 0, 1);
    v16h a0 = load_A16x32(A, Dm, m0, k0);
    v16h a1 = load_A16x32(A, Dm, m0 + 16, k0);
#pragma unroll
    for (int j = 0; j < 4; ++j) {
      v16h b = load_BT16(BT, Dm, n0 + j * 16, k0);
      acc[0][j] = WMMA_F16(a0, b, acc[0][j]);
      acc[1][j] = WMMA_F16(a1, b, acc[1][j]);
    }
  }
#pragma unroll
  for (int i = 0; i < 2; ++i) {
#pragma unroll
    for (int j = 0; j < 4; ++j) {
#pragma unroll
      for (int r = 0; r < 8; ++r) {
        const int ng = n0 + j * 16 + nl;
        const int mg = m0 + i * 16 + r + hi * 8;
        Out[(size_t)mg * Dm + ng] = acc[i][j][r] + bias[ng];
      }
    }
  }
}

// ---------------------------------------------------------------------------
extern "C" void kernel_launch(void* const* d_in, const int* in_sizes, int n_in,
                              void* d_out, int out_size, void* d_ws, size_t ws_size,
                              hipStream_t stream) {
  (void)in_sizes; (void)n_in; (void)out_size; (void)ws_size;
  const float* normed = (const float*)d_in[0];
  // d_in[1] = attn_mask: band computed analytically (identical 0/-1e9 bias)
  const float* Wqkv = (const float*)d_in[2];
  const float* bqkv = (const float*)d_in[3];
  const float* Wout = (const float*)d_in[4];
  const float* bout = (const float*)d_in[5];
  float* out = (float*)d_out;

  char* ws = (char*)d_ws;
  size_t off = 0;
  auto alloc = [&](size_t bytes) -> void* {
    void* p = ws + off;
    off = (off + bytes + 255) & ~(size_t)255;
    return p;
  };
  _Float16* normed_h = (_Float16*)alloc((size_t)Mrows * Dm * 2);       // 8 MB
  _Float16* WqkvT    = (_Float16*)alloc((size_t)N3 * Dm * 2);          // 6 MB
  _Float16* WoutT    = (_Float16*)alloc((size_t)Dm * Dm * 2);          // 2 MB
  _Float16* Qh       = (_Float16*)alloc((size_t)Bsz * H * S * HD * 2); // 8 MB
  _Float16* Kh       = (_Float16*)alloc((size_t)Bsz * H * S * HD * 2); // 8 MB
  _Float16* Vt       = (_Float16*)alloc((size_t)Bsz * H * S * HD * 2); // 8 MB
  _Float16* Mix      = (_Float16*)alloc((size_t)Mrows * Dm * 2);       // 8 MB

  const int n1 = Mrows * Dm;
  k_f32_to_f16<<<(n1 + 255) / 256, 256, 0, stream>>>(normed, normed_h, n1);
  const int n2 = Dm * N3;
  k_transpose_f16<<<(n2 + 255) / 256, 256, 0, stream>>>(Wqkv, WqkvT, Dm, N3);
  const int n3 = Dm * Dm;
  k_transpose_f16<<<(n3 + 255) / 256, 256, 0, stream>>>(Wout, WoutT, Dm, Dm);

  k_qkv_gemm<<<dim3(Mrows / 32, N3 / 64), 32, 0, stream>>>(
      normed_h, WqkvT, bqkv, Qh, Kh, Vt);
  k_attn<<<Bsz * H * (S / 16), 32, 0, stream>>>(Qh, Kh, Vt, Mix);
  k_out_gemm<<<dim3(Mrows / 32, Dm / 64), 32, 0, stream>>>(Mix, WoutT, bout, out);
}
